// PointNet2_32512902431506
// MI455X (gfx1250) — matrix operand order repaired and checked by reading, not compiled, not run on hardware
//
#include <hip/hip_runtime.h>
#include <hip/hip_bf16.h>
#include <stdint.h>

#define BN_EPS 1e-5f

typedef __attribute__((ext_vector_type(16))) __bf16 v16bf;
typedef __attribute__((ext_vector_type(8)))  __bf16 v8bf;
typedef __attribute__((ext_vector_type(8)))  float  v8f;

// ---------------------------------------------------------------------------
// WMMA GEMM:  Y[M,N] = X[M,Kpad] * W[N,Kpad]^T
// X and W are pre-staged bf16, K padded to a multiple of 32 -> single
// branch-free inner loop, fragments loaded as aligned 16B/32B vectors.
// Fragment layouts: 16-bit A 16x32, B 32x16, both with k = 16*half + s (contiguous), f32 C/D.
// ---------------------------------------------------------------------------
__global__ __launch_bounds__(256) void gemm_bf16_wmma(
    const __bf16* __restrict__ X, const __bf16* __restrict__ W,
    float* __restrict__ Y, int M, int Kpad, int N)
{
  const int wave  = threadIdx.x >> 5;
  const int lane  = threadIdx.x & 31;
  const int mtile = blockIdx.x * 8 + wave;
  const int ntile = blockIdx.y;
  if (mtile * 16 >= M) return;            // uniform per wave -> EXEC stays full
  const int half = lane >> 4;             // 0: lanes 0-15, 1: lanes 16-31
  const int l15  = lane & 15;
  const int row  = mtile * 16 + l15;      // A-matrix M index for this lane
  const int col  = ntile * 16 + l15;      // output channel (B-matrix N index)
  const __bf16* __restrict__ xr = X + (long)row * Kpad + half * 16;  // A K-offset
  const __bf16* __restrict__ wr = W + (long)col * Kpad + half * 16;  // B K-offset

  v8f c = {0.f,0.f,0.f,0.f,0.f,0.f,0.f,0.f};
  for (int k0 = 0; k0 < Kpad; k0 += 32) {
    v8bf alo = *(const v8bf*)(xr + k0);        // K chunk [kA, kA+7]
    v8bf ahi = *(const v8bf*)(xr + k0 + 8);    // K chunk [kA+8, kA+15]
    v16bf b  = *(const v16bf*)(wr + k0);       // K chunk [kB, kB+15]
    v16bf a;
#pragma unroll
    for (int i = 0; i < 8; ++i) { a[i] = alo[i]; a[8 + i] = ahi[i]; }
    c = __builtin_amdgcn_wmma_f32_16x16x32_bf16(false, a, false, b,
                                                (short)0, c, false, false);
  }
  const int mBase = mtile * 16 + half * 8;     // C/D layout: vgpr i -> M = mBase+i
#pragma unroll
  for (int i = 0; i < 8; ++i)
    Y[(long)(mBase + i) * N + col] = c[i];
}

// One-time repack of fp32 weights [N,K] -> zero-padded bf16 [N,Kpad].
__global__ void repack_w(const float* __restrict__ W, __bf16* __restrict__ Wb,
                         int K, int Kpad, long total)
{
  long i = (long)blockIdx.x * blockDim.x + threadIdx.x;
  if (i >= total) return;
  int k = (int)(i % Kpad); long n = i / Kpad;
  Wb[i] = (k < K) ? (__bf16)W[n * K + k] : (__bf16)0.f;
}

// ---------------------------------------------------------------------------
// BatchNorm statistics: deterministic two-stage reduction (no float atomics).
// ---------------------------------------------------------------------------
__global__ __launch_bounds__(256) void bn_partial(
    const float* __restrict__ Y, float* __restrict__ psum,
    float* __restrict__ psq, int M, int N, int rowsPerBlock)
{
  const int col  = blockIdx.x * 64 + (threadIdx.x & 63);
  const int rgrp = threadIdx.x >> 6;                  // 0..3
  const int r0 = blockIdx.y * rowsPerBlock;
  int r1 = r0 + rowsPerBlock; if (r1 > M) r1 = M;
  float s = 0.f, s2 = 0.f;
  for (int r = r0 + rgrp; r < r1; r += 4) {
    float v = Y[(long)r * N + col];
    s += v; s2 += v * v;
  }
  __shared__ float ls[4][64], lq[4][64];
  ls[rgrp][threadIdx.x & 63] = s;
  lq[rgrp][threadIdx.x & 63] = s2;
  __syncthreads();
  if (rgrp == 0) {
    int c = threadIdx.x;
    psum[(long)blockIdx.y * N + col] = ls[0][c] + ls[1][c] + ls[2][c] + ls[3][c];
    psq [(long)blockIdx.y * N + col] = lq[0][c] + lq[1][c] + lq[2][c] + lq[3][c];
  }
}

__global__ void bn_finalize(const float* __restrict__ psum,
                            const float* __restrict__ psq,
                            const float* __restrict__ gamma,
                            const float* __restrict__ beta,
                            float* __restrict__ scale, float* __restrict__ shift,
                            int N, int nBlocks, int M)
{
  int c = blockIdx.x * 256 + threadIdx.x;
  if (c >= N) return;
  float s = 0.f, s2 = 0.f;
  for (int b = 0; b < nBlocks; ++b) { s += psum[(long)b * N + c]; s2 += psq[(long)b * N + c]; }
  float invM = 1.f / (float)M;
  float mean = s * invM;
  float var  = s2 * invM - mean * mean;
  if (var < 0.f) var = 0.f;
  float sc = gamma[c] * rsqrtf(var + BN_EPS);
  scale[c] = sc;
  shift[c] = beta[c] - mean * sc;
}

// Intermediate layers: BN+ReLU fp32 -> bf16 next-layer input (N is x32).
__global__ void bn_apply_relu_bf(const float* __restrict__ Y,
                                 __bf16* __restrict__ Xn,
                                 const float* __restrict__ scale,
                                 const float* __restrict__ shift,
                                 long total, int N)
{
  long i = (long)blockIdx.x * blockDim.x + threadIdx.x;
  if (i >= total) return;
  int c = (int)(i % N);
  float v = Y[i] * scale[c] + shift[c];
  Xn[i] = (__bf16)(v > 0.f ? v : 0.f);
}

// Final layer of each MLP: BN+ReLU in place, fp32.
__global__ void bn_apply_relu_f32(float* __restrict__ Y,
                                  const float* __restrict__ scale,
                                  const float* __restrict__ shift,
                                  long total, int N)
{
  long i = (long)blockIdx.x * blockDim.x + threadIdx.x;
  if (i >= total) return;
  int c = (int)(i % N);
  float v = Y[i] * scale[c] + shift[c];
  Y[i] = v > 0.f ? v : 0.f;
}

// ---------------------------------------------------------------------------
// Max-pool over the sample axis: out[g,c] = max_k Y[(g*K+k)*N + c]
// ---------------------------------------------------------------------------
__global__ __launch_bounds__(256) void maxpool_k(
    const float* __restrict__ Y, float* __restrict__ out, int Ksamp, int N)
{
  const int g   = blockIdx.x;
  const int col = blockIdx.y * 64 + (threadIdx.x & 63);
  const int kg  = threadIdx.x >> 6;
  float m = -3.0e38f;
  for (int k = kg; k < Ksamp; k += 4)
    m = fmaxf(m, Y[((long)g * Ksamp + k) * N + col]);
  __shared__ float ls[4][64];
  ls[kg][threadIdx.x & 63] = m;
  __syncthreads();
  if (kg == 0) {
    int c = threadIdx.x;
    out[(long)g * N + col] =
        fmaxf(fmaxf(ls[0][c], ls[1][c]), fmaxf(ls[2][c], ls[3][c]));
  }
}

// Split x (B,N,6) -> xyz (B,N,3), feat (B,N,3)
__global__ void split_input(const float* __restrict__ x, float* __restrict__ xyz,
                            float* __restrict__ feat, int BN)
{
  int i = blockIdx.x * blockDim.x + threadIdx.x;
  if (i >= BN) return;
  xyz [i*3+0] = x[i*6+0]; xyz [i*3+1] = x[i*6+1]; xyz [i*3+2] = x[i*6+2];
  feat[i*3+0] = x[i*6+3]; feat[i*3+1] = x[i*6+4]; feat[i*3+2] = x[i*6+5];
}

// ---------------------------------------------------------------------------
// Farthest point sampling: one block per batch, running min-dist in LDS,
// deterministic first-index argmax (matches jnp.argmax tie-breaking).
// ---------------------------------------------------------------------------
__global__ __launch_bounds__(256) void fps_kernel(
    const float* __restrict__ xyz, int* __restrict__ fpsIdx,
    float* __restrict__ newXyz, int N, int npoint)
{
  const int b = blockIdx.x;
  const float* __restrict__ p = xyz + (long)b * N * 3;
  extern __shared__ float sm[];
  float* dist = sm;                 // N floats
  float* redV = sm + N;             // 256
  int*   redI = (int*)(redV + 256); // 256
  const int t = threadIdx.x;
  for (int i = t; i < N; i += 256) dist[i] = 1e10f;
  __syncthreads();
  int far = 0;
  for (int s = 0; s < npoint; ++s) {
    if (t == 0) {
      fpsIdx[b * npoint + s] = far;
      newXyz[(b * npoint + s) * 3 + 0] = p[far * 3 + 0];
      newXyz[(b * npoint + s) * 3 + 1] = p[far * 3 + 1];
      newXyz[(b * npoint + s) * 3 + 2] = p[far * 3 + 2];
    }
    float cx = p[far*3+0], cy = p[far*3+1], cz = p[far*3+2];
    float bestV = -1.f; int bestI = 0;
    for (int i = t; i < N; i += 256) {
      float dx = p[i*3+0] - cx, dy = p[i*3+1] - cy, dz = p[i*3+2] - cz;
      float d = dx*dx + dy*dy + dz*dz;
      float dm = fminf(dist[i], d);
      dist[i] = dm;
      if (dm > bestV) { bestV = dm; bestI = i; }
    }
    redV[t] = bestV; redI[t] = bestI;
    __syncthreads();
    for (int off = 128; off; off >>= 1) {
      if (t < off) {
        float v2 = redV[t + off]; int i2 = redI[t + off];
        if (v2 > redV[t] || (v2 == redV[t] && i2 < redI[t])) { redV[t] = v2; redI[t] = i2; }
      }
      __syncthreads();
    }
    far = redI[0];
    __syncthreads();
  }
}

// ---------------------------------------------------------------------------
// Ball query: per-query bounded insertion sort (top-nsample by distance) in
// LDS; out-of-radius entries replaced by nearest index (matches reference).
// ---------------------------------------------------------------------------
__global__ __launch_bounds__(64) void ball_query(
    const float* __restrict__ xyz, const float* __restrict__ newXyz,
    int* __restrict__ outIdx, int B, int N, int S, int nsample, float radius)
{
  extern __shared__ float sm2[];
  float* dl = sm2 + (long)threadIdx.x * nsample;
  int*   il = (int*)(sm2 + (long)64 * nsample) + (long)threadIdx.x * nsample;
  const int q = blockIdx.x * 64 + threadIdx.x;
  if (q >= B * S) return;
  const int b = q / S;
  const float* __restrict__ p = xyz + (long)b * N * 3;
  const float qx = newXyz[q*3], qy = newXyz[q*3+1], qz = newXyz[q*3+2];
  int cnt = 0;
  for (int n = 0; n < N; ++n) {
    float dx = p[n*3+0]-qx, dy = p[n*3+1]-qy, dz = p[n*3+2]-qz;
    float d = dx*dx + dy*dy + dz*dz;
    if (cnt < nsample) {
      int pos = cnt;
      while (pos > 0 && dl[pos-1] > d) { dl[pos] = dl[pos-1]; il[pos] = il[pos-1]; --pos; }
      dl[pos] = d; il[pos] = n; ++cnt;
    } else if (d < dl[nsample-1]) {
      int pos = nsample - 1;
      while (pos > 0 && dl[pos-1] > d) { dl[pos] = dl[pos-1]; il[pos] = il[pos-1]; --pos; }
      dl[pos] = d; il[pos] = n;
    }
  }
  const float r2 = radius * radius;
  const int i0 = il[0];
  for (int j = 0; j < nsample; ++j)
    outIdx[(long)q * nsample + j] = (dl[j] > r2) ? i0 : il[j];
}

// grouped[(b*S+s)*ns+k, :] = [ xyz[idx]-new_xyz , pts[idx] , 0-pad ]  -> bf16
__global__ void group_points_bf(const float* __restrict__ xyz,
                                const float* __restrict__ newXyz,
                                const float* __restrict__ pts,
                                const int* __restrict__ idx,
                                __bf16* __restrict__ out,
                                int B, int N, int S, int ns, int Cp, int Cpad)
{
  long total = (long)B * S * ns * Cpad;
  long i = (long)blockIdx.x * blockDim.x + threadIdx.x;
  if (i >= total) return;
  int c = (int)(i % Cpad); long r = i / Cpad;
  int k = (int)(r % ns); long q = r / ns;       // q = b*S+s
  int b = (int)(q / S);
  float v = 0.f;
  if (c < 3) {
    int id = idx[q * ns + k];
    v = xyz[((long)b * N + id) * 3 + c] - newXyz[q * 3 + c];
  } else if (c < 3 + Cp) {
    int id = idx[q * ns + k];
    v = pts[((long)b * N + id) * Cp + (c - 3)];
  }
  out[i] = (__bf16)v;
}

// group_all (sa3): out[(b*N+n), :] = [ xyz , pts , 0-pad ] -> bf16
__global__ void group_all_bf(const float* __restrict__ xyz,
                             const float* __restrict__ pts,
                             __bf16* __restrict__ out, int B, int N, int Cp, int Cpad)
{
  long total = (long)B * N * Cpad;
  long i = (long)blockIdx.x * blockDim.x + threadIdx.x;
  if (i >= total) return;
  int c = (int)(i % Cpad); long r = i / Cpad;   // r = b*N+n
  float v = 0.f;
  if (c < 3) v = xyz[r * 3 + c];
  else if (c < 3 + Cp) v = pts[r * Cp + (c - 3)];
  out[i] = (__bf16)v;
}

// fp3 (S==1 branch): out = concat(p1, broadcast(p2)) -> bf16 (C1+C2 is x32)
__global__ void fuse_broadcast_bf(const float* __restrict__ p1,
                                  const float* __restrict__ p2,
                                  __bf16* __restrict__ out,
                                  int B, int N1, int C1, int C2)
{
  const int C = C1 + C2;
  long total = (long)B * N1 * C;
  long i = (long)blockIdx.x * blockDim.x + threadIdx.x;
  if (i >= total) return;
  int c = (int)(i % C); long q = i / C;         // q = b*N1+n
  int b = (int)(q / N1);
  float v = (c < C1) ? p1[q * C1 + c] : p2[(long)b * C2 + (c - C1)];
  out[i] = (__bf16)v;
}

// 3-NN + inverse-distance weights (first-occurrence tie-break like top_k)
__global__ void knn3(const float* __restrict__ xyz1, const float* __restrict__ xyz2,
                     int* __restrict__ kIdx, float* __restrict__ kW,
                     int B, int N1, int N2)
{
  int q = blockIdx.x * blockDim.x + threadIdx.x;
  if (q >= B * N1) return;
  int b = q / N1;
  const float* __restrict__ p2 = xyz2 + (long)b * N2 * 3;
  float qx = xyz1[q*3], qy = xyz1[q*3+1], qz = xyz1[q*3+2];
  float d0 = 3e38f, d1 = 3e38f, d2 = 3e38f; int i0 = 0, i1 = 0, i2 = 0;
  for (int n = 0; n < N2; ++n) {
    float dx = p2[n*3+0]-qx, dy = p2[n*3+1]-qy, dz = p2[n*3+2]-qz;
    float d = dx*dx + dy*dy + dz*dz;
    if (d < d0)      { d2 = d1; i2 = i1; d1 = d0; i1 = i0; d0 = d; i0 = n; }
    else if (d < d1) { d2 = d1; i2 = i1; d1 = d;  i1 = n; }
    else if (d < d2) { d2 = d;  i2 = n; }
  }
  float w0 = 1.f / (sqrtf(d0) + 1e-8f);
  float w1 = 1.f / (sqrtf(d1) + 1e-8f);
  float w2 = 1.f / (sqrtf(d2) + 1e-8f);
  float inv = 1.f / (w0 + w1 + w2);
  kIdx[q*3+0] = b * N2 + i0; kIdx[q*3+1] = b * N2 + i1; kIdx[q*3+2] = b * N2 + i2;
  kW[q*3+0] = w0 * inv; kW[q*3+1] = w1 * inv; kW[q*3+2] = w2 * inv;
}

// out = concat(p1, interp(p2), 0-pad) -> bf16
__global__ void fuse_interp_bf(const float* __restrict__ p1, const float* __restrict__ p2,
                               const int* __restrict__ kIdx, const float* __restrict__ kW,
                               __bf16* __restrict__ out,
                               int B, int N1, int C1, int C2, int Cpad)
{
  long total = (long)B * N1 * Cpad;
  long i = (long)blockIdx.x * blockDim.x + threadIdx.x;
  if (i >= total) return;
  int c = (int)(i % Cpad); long q = i / Cpad;
  float v = 0.f;
  if (c < C1) v = p1[q * C1 + c];
  else if (c < C1 + C2) {
    int cc = c - C1;
    v = kW[q*3+0] * p2[(long)kIdx[q*3+0] * C2 + cc]
      + kW[q*3+1] * p2[(long)kIdx[q*3+1] * C2 + cc]
      + kW[q*3+2] * p2[(long)kIdx[q*3+2] * C2 + cc];
  }
  out[i] = (__bf16)v;
}

// ---------------------------------------------------------------------------
// Host orchestration
// ---------------------------------------------------------------------------
struct LayerP {
  const float* W; const float* gamma; const float* beta;
  int inC, Kpad, outC; __bf16* wbf;
};
static inline int round32(int k) { return (k + 31) & ~31; }

extern "C" void kernel_launch(void* const* d_in, const int* in_sizes, int n_in,
                              void* d_out, int out_size, void* d_ws, size_t ws_size,
                              hipStream_t stream)
{
  (void)in_sizes; (void)n_in; (void)out_size;
  const int B = 8, N0 = 16384;

  // ---- parse inputs: x, then per layer {W,b,gamma,beta} in dict order ----
  const float* x = (const float*)d_in[0];
  int pi = 1;
  auto L = [&](int inC, int outC) -> LayerP {
    LayerP l; l.W = (const float*)d_in[pi]; l.gamma = (const float*)d_in[pi+2];
    l.beta = (const float*)d_in[pi+3]; l.inC = inC; l.Kpad = round32(inC);
    l.outC = outC; l.wbf = nullptr; pi += 4; return l;
  };
  LayerP sa1[3] = { L(6,64),    L(64,64),   L(64,128)   };
  LayerP sa2[3] = { L(131,128), L(128,128), L(128,256)  };
  LayerP sa3[3] = { L(259,256), L(256,512), L(512,1024) };
  LayerP fp3[2] = { L(1280,256),L(256,256) };
  LayerP fp2[2] = { L(384,256), L(256,128) };
  LayerP fp1[3] = { L(131,128), L(128,128), L(128,256)  };
  LayerP* groups[6] = { sa1, sa2, sa3, fp3, fp2, fp1 };
  const int nlayers[6] = { 3, 3, 3, 2, 2, 3 };

  // ---- workspace bump layout ----
  char* base = (char*)d_ws; size_t off = 0;
  auto alloc = [&](size_t bytes) -> void* {
    void* p = base + off; off += (bytes + 255) & ~(size_t)255; return p;
  };
  float* xyz    = (float*)alloc((size_t)B*N0*3*4);
  float* feat   = (float*)alloc((size_t)B*N0*3*4);
  float* l1_xyz = (float*)alloc((size_t)B*512*3*4);
  float* l2_xyz = (float*)alloc((size_t)B*128*3*4);
  float* l1p    = (float*)alloc((size_t)B*512*128*4);
  float* l2p    = (float*)alloc((size_t)B*128*256*4);
  float* l3p    = (float*)alloc((size_t)B*1024*4);
  float* l2fp   = (float*)alloc((size_t)B*128*256*4);
  float* l1fp   = (float*)alloc((size_t)B*512*128*4);
  int*   fpsI1  = (int*)  alloc((size_t)B*512*4);
  int*   fpsI2  = (int*)  alloc((size_t)B*128*4);
  int*   ball1  = (int*)  alloc((size_t)B*512*32*4);
  int*   ball2  = (int*)  alloc((size_t)B*128*64*4);
  int*   knnI   = (int*)  alloc((size_t)B*N0*3*4);
  float* knnW   = (float*)alloc((size_t)B*N0*3*4);
  float* psum   = (float*)alloc((size_t)128*1024*4);
  float* psq    = (float*)alloc((size_t)128*1024*4);
  float* bnSc   = (float*)alloc((size_t)1024*4);
  float* bnSh   = (float*)alloc((size_t)1024*4);
  // bf16 weight repack region (per layer, padded K)
  for (int g = 0; g < 6; ++g)
    for (int i = 0; i < nlayers[g]; ++i) {
      LayerP& l = groups[g][i];
      l.wbf = (__bf16*)alloc((size_t)l.outC * l.Kpad * 2);
    }
  __bf16* Xa = (__bf16*)alloc((size_t)44*1024*1024);   // bf16 activation ping
  __bf16* Xb = (__bf16*)alloc((size_t)44*1024*1024);   // bf16 activation pong
  float*  Yb = (float*) alloc((size_t)64*1024*1024);   // fp32 GEMM output
  if (off > ws_size) return;  // workspace too small; bail rather than corrupt

  // ---- weight repack (tiny, on-stream, deterministic) ----
  for (int g = 0; g < 6; ++g)
    for (int i = 0; i < nlayers[g]; ++i) {
      LayerP& l = groups[g][i];
      long tot = (long)l.outC * l.Kpad;
      repack_w<<<(unsigned)((tot + 255)/256), 256, 0, stream>>>(l.W, l.wbf,
                                                                l.inC, l.Kpad, tot);
    }

  // ---- GEMM + BN + ReLU layer runner ----
  auto runLayer = [&](const __bf16* Xin, float* Yt, __bf16* Xnext,
                      int M, const LayerP& l) {
    dim3 g1(M / 128, l.outC / 16);
    gemm_bf16_wmma<<<g1, 256, 0, stream>>>(Xin, l.wbf, Yt, M, l.Kpad, l.outC);
    const int rpb = 1024;
    const int nRB = (M + rpb - 1) / rpb;
    dim3 g2(l.outC / 64, nRB);
    bn_partial<<<g2, 256, 0, stream>>>(Yt, psum, psq, M, l.outC, rpb);
    bn_finalize<<<(l.outC + 255)/256, 256, 0, stream>>>(psum, psq, l.gamma, l.beta,
                                                        bnSc, bnSh, l.outC, nRB, M);
    long total = (long)M * l.outC;
    unsigned nb = (unsigned)((total + 255) / 256);
    if (Xnext)
      bn_apply_relu_bf<<<nb, 256, 0, stream>>>(Yt, Xnext, bnSc, bnSh, total, l.outC);
    else
      bn_apply_relu_f32<<<nb, 256, 0, stream>>>(Yt, bnSc, bnSh, total, l.outC);
  };
  // Staged input always lands in Xa; final fp32 result lands in `lastOut`.
  auto runMLP = [&](int M, LayerP* ls, int nL, float* lastOut) {
    const __bf16* cur = Xa;
    for (int i = 0; i < nL; ++i) {
      bool last = (i == nL - 1);
      __bf16* nxt = last ? nullptr : ((cur == Xa) ? Xb : Xa);
      runLayer(cur, last ? lastOut : Yb, nxt, M, ls[i]);
      cur = nxt;
    }
  };

  // ---- stage 0: split ----
  split_input<<<(B*N0 + 255)/256, 256, 0, stream>>>(x, xyz, feat, B*N0);

  // ---- SA1: 16384 -> 512, r=0.1, ns=32, [64,64,128] ----
  fps_kernel<<<B, 256, (size_t)N0*4 + 256*8, stream>>>(xyz, fpsI1, l1_xyz, N0, 512);
  ball_query<<<(B*512 + 63)/64, 64, (size_t)64*32*8, stream>>>(
      xyz, l1_xyz, ball1, B, N0, 512, 32, 0.1f);
  {
    long tot = (long)B*512*32*32;   // Cpad(6) = 32
    group_points_bf<<<(unsigned)((tot + 255)/256), 256, 0, stream>>>(
        xyz, l1_xyz, feat, ball1, Xa, B, N0, 512, 32, 3, 32);
  }
  runMLP(B*512*32, sa1, 3, Yb);
  maxpool_k<<<dim3(B*512, 128/64), 256, 0, stream>>>(Yb, l1p, 32, 128);

  // ---- SA2: 512 -> 128, r=0.2, ns=64, [128,128,256] ----
  fps_kernel<<<B, 256, (size_t)512*4 + 256*8, stream>>>(l1_xyz, fpsI2, l2_xyz, 512, 128);
  ball_query<<<(B*128 + 63)/64, 64, (size_t)64*64*8, stream>>>(
      l1_xyz, l2_xyz, ball2, B, 512, 128, 64, 0.2f);
  {
    long tot = (long)B*128*64*160;  // Cpad(131) = 160
    group_points_bf<<<(unsigned)((tot + 255)/256), 256, 0, stream>>>(
        l1_xyz, l2_xyz, l1p, ball2, Xa, B, 512, 128, 64, 128, 160);
  }
  runMLP(B*128*64, sa2, 3, Yb);
  maxpool_k<<<dim3(B*128, 256/64), 256, 0, stream>>>(Yb, l2p, 64, 256);

  // ---- SA3: group_all over 128 pts, [256,512,1024] ----
  {
    long tot = (long)B*128*288;     // Cpad(259) = 288
    group_all_bf<<<(unsigned)((tot + 255)/256), 256, 0, stream>>>(
        l2_xyz, l2p, Xa, B, 128, 256, 288);
  }
  runMLP(B*128, sa3, 3, Yb);
  maxpool_k<<<dim3(B, 1024/64), 256, 0, stream>>>(Yb, l3p, 128, 1024);

  // ---- FP3: broadcast l3 -> 128 pts, concat l2 (256+1024=1280, x32), [256,256] ----
  {
    long tot = (long)B*128*1280;
    fuse_broadcast_bf<<<(unsigned)((tot + 255)/256), 256, 0, stream>>>(
        l2p, l3p, Xa, B, 128, 256, 1024);
  }
  runMLP(B*128, fp3, 2, l2fp);

  // ---- FP2: 3-NN interp 128 -> 512, concat l1 (128+256=384, x32), [256,128] ----
  knn3<<<(B*512 + 255)/256, 256, 0, stream>>>(l1_xyz, l2_xyz, knnI, knnW, B, 512, 128);
  {
    long tot = (long)B*512*384;
    fuse_interp_bf<<<(unsigned)((tot + 255)/256), 256, 0, stream>>>(
        l1p, l2fp, knnI, knnW, Xa, B, 512, 128, 256, 384);
  }
  runMLP(B*512, fp2, 2, l1fp);

  // ---- FP1: 3-NN interp 512 -> 16384, concat feat (3+128=131 -> pad 160) ----
  knn3<<<(B*N0 + 255)/256, 256, 0, stream>>>(xyz, l1_xyz, knnI, knnW, B, N0, 512);
  {
    long tot = (long)B*N0*160;      // Cpad(131) = 160
    fuse_interp_bf<<<(unsigned)((tot + 255)/256), 256, 0, stream>>>(
        feat, l1fp, knnI, knnW, Xa, B, N0, 3, 128, 160);
  }
  runMLP(B*N0, fp1, 3, (float*)d_out);   // final BN+ReLU lands in d_out
}
